// LinearAttention_52836687676079
// MI455X (gfx1250) — matrix operand order repaired and checked by reading, not compiled, not run on hardware
//
#include <hip/hip_runtime.h>
#include <hip/hip_bf16.h>

// Chunked causal linear attention for gfx1250 (MI455X).
// out = tril(q k^T) v  ==  per chunk:  out_c = q_c @ S_prev + tril(q_c k_c^T) @ v_c ;  S += k_c^T v_c
// bf16 WMMA (v_wmma_f32_16x16x32_bf16), fp32 accumulation, fp32 running state in VGPRs.
// All LDS operand layouts are chosen so every WMMA fragment load is contiguous (ds_load_b128 pairs).

typedef __attribute__((ext_vector_type(16))) __bf16 v16bf;
typedef __attribute__((ext_vector_type(8)))  __bf16 v8bf;
typedef __attribute__((ext_vector_type(4)))  __bf16 v4bf;
typedef __attribute__((ext_vector_type(8)))  float  v8f;

#define DEV static __device__ __forceinline__

// A operand, 16x32 bf16, row-major source src[row][col], leading dim ld.
// lanes 0-15: M=lane, K=0..15; lanes 16-31: K=16..31 (contiguous, matches load_B_nmajor).
// Elements 0..15 (+16h) are contiguous -> two ds_load_b128.
DEV v16bf load_A_rowmajor(const __bf16* src, int ld, int row0, int col0, int lane) {
  const int half = lane >> 4, m = lane & 15;
  const __bf16* base = src + (row0 + m) * ld + col0;
  v16bf a;
  #pragma unroll
  for (int r = 0; r < 8; ++r) {
    const int k0 = 2 * r + 16 * half;
    a[2 * r]     = base[k0];
    a[2 * r + 1] = base[k0 + 1];
  }
  return a;
}

// B operand, 32x16 bf16, from N-major storage srcT[n][k] (i.e. B[k][n] = srcT[n*ld + k]).
// Per lane: fixed N = lane&15 (+0/+16 K offset by half), 16 consecutive K values -> two ds_load_b128.
DEV v16bf load_B_nmajor(const __bf16* srcT, int ld, int k0b, int n0, int lane) {
  const int half = lane >> 4, n = lane & 15;
  const __bf16* base = srcT + (n0 + n) * ld + k0b + 16 * half;
  v16bf b;
  #pragma unroll
  for (int r = 0; r < 16; ++r) b[r] = base[r];
  return b;
}

DEV v8f wmma_bf16(v16bf a, v16bf b, v8f c) {
  return __builtin_amdgcn_wmma_f32_16x16x32_bf16(false, a, false, b, (short)0, c, false, false);
}

DEV v4bf cvt4(const float4 f) {
  return (v4bf){(__bf16)f.x, (__bf16)f.y, (__bf16)f.z, (__bf16)f.w};
}

__global__ __launch_bounds__(256)
void LinearAttention_52836687676079_kernel(const float* __restrict__ q,
                                           const float* __restrict__ k,
                                           const float* __restrict__ v,
                                           float* __restrict__ o) {
  constexpr int T = 2048, D = 128, C = 64, NCH = T / C;
  const int head = blockIdx.x;          // b*H + h, 0..63
  const int tid  = threadIdx.x;
  const int lane = tid & 31;
  const int wave = tid >> 5;            // 0..7 : owns output/state columns [wave*16, wave*16+16)
  const int half = lane >> 4, nlo = lane & 15;
  const size_t base = (size_t)head * T * D;

  extern __shared__ __bf16 smem[];
  __bf16* q_s  = smem;                  // [C][D] row-major   (A operand: phases A, B1)
  __bf16* k_s  = q_s  + C * D;          // [C][D] row-major   (B operand k^T via N-major read: B1)
  __bf16* k_sT = k_s  + C * D;          // [D][C] transposed  (A operand k^T: phase D)
  __bf16* v_sT = k_sT + D * C;          // [D][C] transposed  (B operand v: phases B2, D)
  __bf16* S_sT = v_sT + D * C;          // [D][D] d_col-major (B operand S: phase A)
  __bf16* sc_s = S_sT + D * D;          // [C][C] row-major   (A operand scores: phase B2)

  for (int i = tid; i < D * D; i += 256) S_sT[i] = (__bf16)0.f;

  // fp32 running state S[:, wave*16:+16] as 8 accumulator tiles (di = 0..7)
  v8f Sacc[8];
  #pragma unroll
  for (int i = 0; i < 8; ++i) Sacc[i] = (v8f){0, 0, 0, 0, 0, 0, 0, 0};

  // staging geometry: each thread owns 32 contiguous elements of the 64x128 chunk
  const int jr = tid >> 2;              // row 0..63
  const int d0 = (tid & 3) * 32;        // col block
  const int i0 = jr * D + d0;           // == tid*32

  __syncthreads();

  for (int t = 0; t < NCH; ++t) {
    const size_t cbase = base + (size_t)t * C * D;

    // ---- stage q,k,v chunk to LDS as bf16 ----
    #pragma unroll
    for (int dd = 0; dd < 32; dd += 4) {
      const float4 fq = *(const float4*)(q + cbase + i0 + dd);
      *(v4bf*)(q_s + i0 + dd) = cvt4(fq);

      const float4 fk = *(const float4*)(k + cbase + i0 + dd);
      const v4bf kb = cvt4(fk);
      *(v4bf*)(k_s + i0 + dd) = kb;                 // row-major copy
      k_sT[(d0 + dd + 0) * C + jr] = kb[0];         // transposed copy
      k_sT[(d0 + dd + 1) * C + jr] = kb[1];
      k_sT[(d0 + dd + 2) * C + jr] = kb[2];
      k_sT[(d0 + dd + 3) * C + jr] = kb[3];

      const float4 fv = *(const float4*)(v + cbase + i0 + dd);
      const v4bf vb = cvt4(fv);
      v_sT[(d0 + dd + 0) * C + jr] = vb[0];         // transposed only
      v_sT[(d0 + dd + 1) * C + jr] = vb[1];
      v_sT[(d0 + dd + 2) * C + jr] = vb[2];
      v_sT[(d0 + dd + 3) * C + jr] = vb[3];
    }
    // prefetch next chunk (global_prefetch_b8), 128B line per thread per tensor
    if (t + 1 < NCH) {
      const size_t nb = cbase + (size_t)C * D + (size_t)tid * 32;
      __builtin_prefetch(q + nb, 0, 1);
      __builtin_prefetch(k + nb, 0, 1);
      __builtin_prefetch(v + nb, 0, 1);
    }
    __syncthreads();

    // ---- Phase A: oacc(ci, wave) = q_chunk @ S_prev  (inter-chunk causal part) ----
    v8f oacc[4];
    #pragma unroll
    for (int ci = 0; ci < 4; ++ci) oacc[ci] = (v8f){0, 0, 0, 0, 0, 0, 0, 0};
    #pragma unroll
    for (int di = 0; di < 4; ++di) {                 // K over D in steps of 32
      const v16bf b = load_B_nmajor(S_sT, D, di * 32, wave * 16, lane);
      #pragma unroll
      for (int ci = 0; ci < 4; ++ci) {
        const v16bf a = load_A_rowmajor(q_s, D, ci * 16, di * 32, lane);
        oacc[ci] = wmma_bf16(a, b, oacc[ci]);
      }
    }

    // ---- Phase B1: scores = tril(q_c k_c^T), bf16 into LDS. 16 tiles / 8 waves ----
    #pragma unroll
    for (int s = 0; s < 2; ++s) {
      const int tIdx = wave + 8 * s;                 // wave-uniform
      const int ci = tIdx >> 2, cj = tIdx & 3;
      if (cj > ci) {                                 // strictly upper tile: all zero
        #pragma unroll
        for (int r = 0; r < 8; ++r)
          sc_s[(ci * 16 + r + 8 * half) * C + cj * 16 + nlo] = (__bf16)0.f;
      } else {
        v8f acc = {0, 0, 0, 0, 0, 0, 0, 0};
        #pragma unroll
        for (int kd = 0; kd < 4; ++kd) {             // K over D
          const v16bf a = load_A_rowmajor(q_s, D, ci * 16, kd * 32, lane);
          const v16bf b = load_B_nmajor(k_s, D, kd * 32, cj * 16, lane); // k^T: N-major == row-major k
          acc = wmma_bf16(a, b, acc);
        }
        if (ci == cj) {                              // diagonal tile: mask j>i
          #pragma unroll
          for (int r = 0; r < 8; ++r)
            if (nlo > r + 8 * half) acc[r] = 0.f;
        }
        #pragma unroll
        for (int r = 0; r < 8; ++r)
          sc_s[(ci * 16 + r + 8 * half) * C + cj * 16 + nlo] = (__bf16)acc[r];
      }
    }
    __syncthreads();

    // v fragments (shared by Phase B2 and Phase D)
    v16bf bv[2];
    #pragma unroll
    for (int jb = 0; jb < 2; ++jb)
      bv[jb] = load_B_nmajor(v_sT, C, jb * 32, wave * 16, lane);

    // ---- Phase B2: oacc += scores @ v_chunk  (intra-chunk causal part) ----
    #pragma unroll
    for (int jb = 0; jb < 2; ++jb) {
      #pragma unroll
      for (int ci = 2 * jb; ci < 4; ++ci) {          // tiles with cj>ci are zero: skip
        const v16bf a = load_A_rowmajor(sc_s, C, ci * 16, jb * 32, lane);
        oacc[ci] = wmma_bf16(a, bv[jb], oacc[ci]);
      }
    }

    // ---- Phase C: write out chunk (fp32), coalesced across N within half-wave ----
    #pragma unroll
    for (int ci = 0; ci < 4; ++ci) {
      #pragma unroll
      for (int r = 0; r < 8; ++r) {
        const int row = t * C + ci * 16 + r + 8 * half;
        o[base + (size_t)row * D + wave * 16 + nlo] = oacc[ci][r];
      }
    }

    // ---- Phase D: S(:, wave) += k_c^T @ v_c ; refresh bf16 N-major copy in LDS ----
    #pragma unroll
    for (int jb = 0; jb < 2; ++jb) {                 // K over C
      #pragma unroll
      for (int di = 0; di < 8; ++di) {
        const v16bf a = load_A_rowmajor(k_sT, C, di * 16, jb * 32, lane);
        Sacc[di] = wmma_bf16(a, bv[jb], Sacc[di]);
      }
    }
    #pragma unroll
    for (int di = 0; di < 8; ++di) {                 // one ds_store_b128 per tile
      v8bf p;
      #pragma unroll
      for (int r = 0; r < 8; ++r) p[r] = (__bf16)Sacc[di][r];
      *(v8bf*)(S_sT + (wave * 16 + nlo) * D + di * 16 + 8 * half) = p;
    }
    __syncthreads();                                 // S_sT / k / v safe for next iter
  }
}

extern "C" void kernel_launch(void* const* d_in, const int* in_sizes, int n_in,
                              void* d_out, int out_size, void* d_ws, size_t ws_size,
                              hipStream_t stream) {
  const float* q = (const float*)d_in[0];
  const float* k = (const float*)d_in[1];
  const float* v = (const float*)d_in[2];
  float* o = (float*)d_out;

  constexpr int BH = 4 * 16;
  // q_s + k_s + k_sT + v_sT + S_sT + sc_s  (bf16 elements)
  constexpr size_t SHMEM =
      (size_t)(64 * 128 * 4 + 128 * 128 + 64 * 64) * 2;  // 106496 bytes

  hipLaunchKernelGGL(LinearAttention_52836687676079_kernel,
                     dim3(BH), dim3(256), SHMEM, stream, q, k, v, o);
}
